// QLSTMPlus_65481071404458
// MI455X (gfx1250) — compile-verified, matrix operand order checked by reading, not run on hardware
//
#include <hip/hip_runtime.h>
#include <hip/hip_bf16.h>

// ---------------------------------------------------------------------------
// Reference: degenerate 2-layer LSTM (h_prev==0 every step) ->
// per layer: 3 GEMMs sharing A (M=S*B=262144, K=D=256, N=H=256) for gates
// i,g,o (f is dead: multiplies zero c_prev), then
//     h = sigmoid(o) * tanh( sigmoid(i) * tanh(g) )
// followed by LayerNorm over H.  bf16 WMMA, f32 accumulation.
//
// Tiling: block = 256 threads (8 wave32), tile = 64 rows x full H=256.
// Wave w owns N columns [32w, 32w+32).  Gates processed as 3 sequential
// K-passes with a running 64-VGPR partial.  kc-loop unrolled only 2x and
// waves/EU >= 4 requested to keep VGPR liveness < 256 (avoid s_set_vgpr_msb
// churn observed with full unroll).
// ---------------------------------------------------------------------------

#define DD    256          // input dim (= K)
#define HH    256          // hidden dim (= N)
#define MTOT  (2048 * 128) // M = S*B
#define MT    64           // rows per block
#define NMT   4            // m-tiles of 16 per wave
#define NGATE 3            // i, g, o (source gates 1,2,3)
#define KC    8            // K chunks of 32
#define NT    16           // N tiles of 16
#define FRAG_PER_GATE (NT * KC * 32 * 16)        // ushorts
#define WB_PER_LAYER  (NGATE * FRAG_PER_GATE)    // 196608 ushorts / layer

typedef __attribute__((ext_vector_type(16))) __bf16 v16bf;
typedef __attribute__((ext_vector_type(8)))  float  v8f;

union Frag32B {            // 32 bytes: one WMMA 16-bit operand fragment / lane
    uint4 q[2];
    v16bf v;
};

__device__ __forceinline__ unsigned short f32_to_bf16_rne(float f) {
    unsigned int u = __float_as_uint(f);
    u += 0x7FFFu + ((u >> 16) & 1u);       // round-to-nearest-even
    return (unsigned short)(u >> 16);
}

__device__ __forceinline__ float fsigmoid(float x) {
    return 1.0f / (1.0f + __expf(-x));
}
__device__ __forceinline__ float ftanh(float x) {
    float e = __expf(2.0f * x);
    return 1.0f - 2.0f / (e + 1.0f);       // safe at +/-inf
}

// ---------------------------------------------------------------------------
// Pre-swizzle weights into WMMA B-fragment layout (bf16).
// Fragment index: ((((l*3+g)*NT + nt)*KC + kc)*32 + lane)*16 + e
// B matrix is 32(K) x 16(N): lane%16 = N; lanes 0-15 hold K {0..7,16..23},
// lanes 16-31 hold K {8..15,24..31} (CDNA5 16-bit operand layout).
// ---------------------------------------------------------------------------
__global__ void pack_weights(const float* __restrict__ W, unsigned short* __restrict__ Wb) {
    int idx = blockIdx.x * blockDim.x + threadIdx.x; // 2 * WB_PER_LAYER threads
    int t = idx;
    int e    = t & 15;  t >>= 4;
    int lane = t & 31;  t >>= 5;
    int kc   = t & 7;   t >>= 3;
    int nt   = t & 15;  t >>= 4;
    int g    = t % 3;
    int l    = t / 3;

    int khalf = lane >> 4;
    int k = kc * 32 + ((e < 8) ? (khalf * 8 + e) : (16 + khalf * 8 + (e - 8)));
    int n = nt * 16 + (lane & 15);
    int gsrc = g + 1;                                   // skip dead f-gate (0)
    // W layout: (L, 4, D+H, H) row-major; only k < D matters (h_prev == 0)
    size_t src = ((size_t)(l * 4 + gsrc) * (DD + HH) + (size_t)k) * HH + n;
    Wb[idx] = f32_to_bf16_rne(W[src]);
}

// ---------------------------------------------------------------------------
// One K=256 GEMM pass for a single gate: acc[mt][nl] = As x Wfrag.
// kc unrolled only 2x: bounds in-flight A fragments (VGPR liveness).
// ---------------------------------------------------------------------------
__device__ __forceinline__ void gemm_pass(const unsigned short* __restrict__ Wfrag,
                                          const unsigned short* As,
                                          int wave, int lane,
                                          v8f acc[NMT][2]) {
    const int khalf = lane >> 4;
    const int nlane = lane & 15;
    #pragma unroll
    for (int mt = 0; mt < NMT; ++mt)
        #pragma unroll
        for (int nl = 0; nl < 2; ++nl)
            acc[mt][nl] = (v8f){0.f,0.f,0.f,0.f,0.f,0.f,0.f,0.f};

    #pragma unroll 2
    for (int kc = 0; kc < KC; ++kc) {
        Frag32B af[NMT];
        #pragma unroll
        for (int mt = 0; mt < NMT; ++mt) {
            int row = mt * 16 + nlane;
            int k0  = kc * 32 + khalf * 8;
            af[mt].q[0] = *(const uint4*)&As[row * DD + k0];
            af[mt].q[1] = *(const uint4*)&As[row * DD + k0 + 16];
        }
        #pragma unroll
        for (int nl = 0; nl < 2; ++nl) {
            int nt = wave * 2 + nl;
            const unsigned short* p =
                Wfrag + (((size_t)nt * KC + kc) * 32 + lane) * 16;
            Frag32B bfr;
            bfr.q[0] = *(const uint4*)p;
            bfr.q[1] = *(const uint4*)(p + 8);
            #pragma unroll
            for (int mt = 0; mt < NMT; ++mt)
                acc[mt][nl] = __builtin_amdgcn_wmma_f32_16x16x32_bf16(
                    false, af[mt].v, false, bfr.v,
                    (short)0, acc[mt][nl], false, false);
        }
    }
}

// ---------------------------------------------------------------------------
// One layer: 3-pass GEMM + activations + LayerNorm.
// ---------------------------------------------------------------------------
__global__ void __launch_bounds__(256, 4)
lstm_layer(const float* __restrict__ A,           // (MTOT, DD) fp32
           const unsigned short* __restrict__ Wb, // this layer's fragments
           const float* __restrict__ bias,        // this layer's (4, H)
           const float* __restrict__ lng,         // (H)
           const float* __restrict__ lnb,         // (H)
           float* __restrict__ out)               // (MTOT, HH) fp32
{
    __shared__ union {
        unsigned short As[MT * DD];   // 32 KB bf16 A tile (GEMM phase)
        float          Hs[MT][HH + 8];// 68 KB h staging (LayerNorm phase)
    } sm;

    const int tid   = threadIdx.x;
    const int wave  = tid >> 5;
    const int lane  = tid & 31;
    const int khalf = lane >> 4;
    const int nlane = lane & 15;
    const size_t m0 = (size_t)blockIdx.x * MT;

    // ---- Stage A tile: fp32 global -> bf16 LDS (each thread 64 elements) ----
    #pragma unroll 4
    for (int i = 0; i < 16; ++i) {
        int e = tid * 64 + i * 4;            // 0..16383, row-contiguous
        int r = e >> 8;
        int c = e & 255;
        float4 v = *(const float4*)(A + (m0 + r) * DD + c);
        unsigned int lo = (unsigned int)f32_to_bf16_rne(v.x) |
                          ((unsigned int)f32_to_bf16_rne(v.y) << 16);
        unsigned int hi = (unsigned int)f32_to_bf16_rne(v.z) |
                          ((unsigned int)f32_to_bf16_rne(v.w) << 16);
        *(uint2*)&sm.As[e] = make_uint2(lo, hi);
    }
    __syncthreads();

    v8f acc[NMT][2];
    v8f held[NMT][2];

    // ---- Pass 0: gate i (src gate 1): held = sigmoid(i) ----
    gemm_pass(Wb + 0 * FRAG_PER_GATE, sm.As, wave, lane, acc);
    #pragma unroll
    for (int nl = 0; nl < 2; ++nl) {
        float bi = bias[1 * HH + (wave * 2 + nl) * 16 + nlane];
        #pragma unroll
        for (int mt = 0; mt < NMT; ++mt)
            #pragma unroll
            for (int r = 0; r < 8; ++r)
                held[mt][nl][r] = fsigmoid(acc[mt][nl][r] + bi);
    }

    // ---- Pass 1: gate g (src gate 2): held = tanh( held * tanh(g) ) ----
    gemm_pass(Wb + 1 * FRAG_PER_GATE, sm.As, wave, lane, acc);
    #pragma unroll
    for (int nl = 0; nl < 2; ++nl) {
        float bg = bias[2 * HH + (wave * 2 + nl) * 16 + nlane];
        #pragma unroll
        for (int mt = 0; mt < NMT; ++mt)
            #pragma unroll
            for (int r = 0; r < 8; ++r) {
                float c = held[mt][nl][r] * ftanh(acc[mt][nl][r] + bg);
                held[mt][nl][r] = ftanh(c);     // tanh(c) for the o-gate mult
            }
    }

    // ---- Pass 2: gate o (src gate 3): h = sigmoid(o) * held ----
    gemm_pass(Wb + 2 * FRAG_PER_GATE, sm.As, wave, lane, acc);
    #pragma unroll
    for (int nl = 0; nl < 2; ++nl) {
        float bo = bias[3 * HH + (wave * 2 + nl) * 16 + nlane];
        #pragma unroll
        for (int mt = 0; mt < NMT; ++mt)
            #pragma unroll
            for (int r = 0; r < 8; ++r)
                held[mt][nl][r] = fsigmoid(acc[mt][nl][r] + bo) * held[mt][nl][r];
    }

    // ---- Alias switch: all As reads done -> stage h rows for LayerNorm ----
    __syncthreads();
    #pragma unroll
    for (int nl = 0; nl < 2; ++nl) {
        int n = (wave * 2 + nl) * 16 + nlane;
        #pragma unroll
        for (int mt = 0; mt < NMT; ++mt)
            #pragma unroll
            for (int r = 0; r < 8; ++r) {
                int m = mt * 16 + r + 8 * khalf;   // C/D layout row
                sm.Hs[m][n] = held[mt][nl][r];
            }
    }
    __syncthreads();

    // ---- LayerNorm over H=256: wave w handles rows 8w..8w+7 ----
    #pragma unroll 2
    for (int rr = 0; rr < 8; ++rr) {
        int m = wave * 8 + rr;
        float4 x0 = *(const float4*)&sm.Hs[m][lane * 8];
        float4 x1 = *(const float4*)&sm.Hs[m][lane * 8 + 4];
        float s  = x0.x + x0.y + x0.z + x0.w + x1.x + x1.y + x1.z + x1.w;
        float ss = x0.x*x0.x + x0.y*x0.y + x0.z*x0.z + x0.w*x0.w
                 + x1.x*x1.x + x1.y*x1.y + x1.z*x1.z + x1.w*x1.w;
        #pragma unroll
        for (int off = 16; off >= 1; off >>= 1) {
            s  += __shfl_xor(s,  off, 32);
            ss += __shfl_xor(ss, off, 32);
        }
        float mu   = s * (1.0f / 256.0f);
        float var  = ss * (1.0f / 256.0f) - mu * mu;
        float rstd = __frsqrt_rn(var + 1e-5f);

        float4 g0 = *(const float4*)&lng[lane * 8];
        float4 g1 = *(const float4*)&lng[lane * 8 + 4];
        float4 b0 = *(const float4*)&lnb[lane * 8];
        float4 b1 = *(const float4*)&lnb[lane * 8 + 4];

        float4 y0, y1;
        y0.x = (x0.x - mu) * rstd * g0.x + b0.x;
        y0.y = (x0.y - mu) * rstd * g0.y + b0.y;
        y0.z = (x0.z - mu) * rstd * g0.z + b0.z;
        y0.w = (x0.w - mu) * rstd * g0.w + b0.w;
        y1.x = (x1.x - mu) * rstd * g1.x + b1.x;
        y1.y = (x1.y - mu) * rstd * g1.y + b1.y;
        y1.z = (x1.z - mu) * rstd * g1.z + b1.z;
        y1.w = (x1.w - mu) * rstd * g1.w + b1.w;

        float* op = out + (m0 + m) * HH + lane * 8;
        *(float4*)op       = y0;
        *(float4*)(op + 4) = y1;
    }
}

// ---------------------------------------------------------------------------
// Host launcher. d_in order: inputs, W, b, ln_g, ln_b (all fp32).
// Layer 0: inputs -> d_out.  Layer 1: d_out -> d_out in place (each block
// stages its own 64 rows to LDS before overwriting them; rows independent).
// ---------------------------------------------------------------------------
extern "C" void kernel_launch(void* const* d_in, const int* in_sizes, int n_in,
                              void* d_out, int out_size, void* d_ws, size_t ws_size,
                              hipStream_t stream) {
    const float* inp = (const float*)d_in[0];
    const float* W   = (const float*)d_in[1];
    const float* b   = (const float*)d_in[2];
    const float* lng = (const float*)d_in[3];
    const float* lnb = (const float*)d_in[4];
    float* out = (float*)d_out;
    unsigned short* Wb = (unsigned short*)d_ws;   // 2 * 196608 ushorts = 786 KB

    // 1) swizzle weights into bf16 WMMA-B fragments (L2-resident thereafter)
    pack_weights<<<(2 * WB_PER_LAYER) / 256, 256, 0, stream>>>(W, Wb);

    const int nblocks = MTOT / MT; // 4096
    // 2) layer 0
    lstm_layer<<<nblocks, 256, 0, stream>>>(inp, Wb,
                                            b,          lng,      lnb,      out);
    // 3) layer 1 (in place)
    lstm_layer<<<nblocks, 256, 0, stream>>>(out, Wb + WB_PER_LAYER,
                                            b + 4 * HH, lng + HH, lnb + HH, out);
}
